// NonLocal_2559800508895
// MI455X (gfx1250) — compile-verified, matrix operand order checked by reading, not run on hardware
//
#include <hip/hip_runtime.h>
#include <hip/hip_bf16.h>
#include <math.h>

typedef __bf16 bf16;
typedef bf16  v16bf __attribute__((ext_vector_type(16)));
typedef bf16  v8bf  __attribute__((ext_vector_type(8)));
typedef float v8f   __attribute__((ext_vector_type(8)));

#define NB   8
#define C    1024
#define D    512
#define L    3136   // 16*14*14
#define P    784    // 16*7*7
#define PB   832    // P padded to multiple of 64
#define RSQRT_D 0.044194173824159216f  // 512^-0.5

// Load a WMMA 16-bit A/B fragment from a row-major [16][K] tile (Bt rows for B).
// Lane l: row = l&15; halves 0..7 = row[k0 + 8*(l>>4) .. +7], halves 8..15 = row[+16].
static __device__ inline v16bf load_frag(const bf16* base, int ld, int lane) {
  const bf16* p = base + (size_t)(lane & 15) * ld + ((lane >> 4) << 3);
  v8bf lo = *(const v8bf*)p;
  v8bf hi = *(const v8bf*)(p + 16);
  return __builtin_shufflevector(lo, hi, 0,1,2,3,4,5,6,7,8,9,10,11,12,13,14,15);
}

static __device__ inline v8f wmma_bf16(v16bf a, v16bf b, v8f c) {
  return __builtin_amdgcn_wmma_f32_16x16x32_bf16(false, a, false, b, (short)0, c,
                                                 false, false);
}

// Wave-level macro-tile GEMM: (MI*16) x (NI*16) output, A rows / Bt rows row-major.
template <int MI, int NI>
static __device__ inline void gemm_tile(const bf16* A, int lda, const bf16* B, int ldb,
                                        int K, int lane, v8f (&acc)[MI][NI]) {
  for (int k = 0; k < K; k += 32) {
    v16bf a[MI], b[NI];
#pragma unroll
    for (int i = 0; i < MI; ++i) a[i] = load_frag(A + (size_t)i * 16 * lda + k, lda, lane);
#pragma unroll
    for (int j = 0; j < NI; ++j) b[j] = load_frag(B + (size_t)j * 16 * ldb + k, ldb, lane);
#pragma unroll
    for (int i = 0; i < MI; ++i)
#pragma unroll
      for (int j = 0; j < NI; ++j) acc[i][j] = wmma_bf16(a[i], b[j], acc[i][j]);
  }
}

// ---------------- packing kernels ----------------

__global__ void cvt_bf16_kernel(const float* __restrict__ in, bf16* __restrict__ out, int n) {
  int i = blockIdx.x * blockDim.x + threadIdx.x;
  int stride = gridDim.x * blockDim.x;
  for (; i < n; i += stride) out[i] = (bf16)in[i];
}

// x [N][C][L] f32 -> xT [N][L][C] bf16 (tiled transpose)
__global__ void xpose_kernel(const float* __restrict__ x, bf16* __restrict__ xT) {
  __shared__ float tile[32][33];
  int n = blockIdx.z, c0 = blockIdx.y * 32, l0 = blockIdx.x * 32;
  int tx = threadIdx.x, ty = threadIdx.y;  // 32 x 8
  const float* xs = x + (size_t)n * C * L;
#pragma unroll
  for (int i = 0; i < 4; ++i)
    tile[ty + i * 8][tx] = xs[(size_t)(c0 + ty + i * 8) * L + (l0 + tx)];
  __syncthreads();
  bf16* xo = xT + (size_t)n * L * C;
#pragma unroll
  for (int i = 0; i < 4; ++i)
    xo[(size_t)(l0 + ty + i * 8) * C + (c0 + tx)] = (bf16)tile[tx][ty + i * 8];
}

// 2x2 max-pool on HxW then transpose: x [N][C][L] f32 -> xpT [N][PB][C] bf16
// (rows [P, PB) are zero-filled so all downstream GEMMs use uniform 64-wide tiles)
__global__ void pool_xpose_kernel(const float* __restrict__ x, bf16* __restrict__ xpT) {
  __shared__ float tile[32][33];
  int n = blockIdx.z, c0 = blockIdx.y * 32, p0 = blockIdx.x * 32;
  int tx = threadIdx.x, ty = threadIdx.y;
  const float* xs = x + (size_t)n * C * L;
#pragma unroll
  for (int i = 0; i < 4; ++i) {
    int c = c0 + ty + i * 8;
    int p = p0 + tx;
    float v = 0.f;
    if (p < P) {
      int t = p / 49, r = p % 49;
      int h = r / 7, w = r % 7;
      const float* b = xs + (size_t)c * L + t * 196 + (2 * h) * 14 + 2 * w;
      v = fmaxf(fmaxf(b[0], b[1]), fmaxf(b[14], b[15]));
    }
    tile[ty + i * 8][tx] = v;
  }
  __syncthreads();
  bf16* xo = xpT + (size_t)n * PB * C;
#pragma unroll
  for (int i = 0; i < 4; ++i) {
    int p = p0 + ty + i * 8;
    xo[(size_t)p * C + (c0 + tx)] = (bf16)tile[tx][ty + i * 8];
  }
}

// ---------------- projection GEMM (64x64 wave tiles) ----------------
// theta/phi (mode 0): out[n][l][D] bf16, contiguous per-lane b128 stores
// g        (mode 1): out[n][d][PB] bf16, strided stores
__global__ void __launch_bounds__(32, 1)
proj_kernel(const bf16* __restrict__ Wt, const bf16* __restrict__ xT,
            bf16* __restrict__ out, int Lsrc, int mode) {
  int lane = threadIdx.x;  // one wave32 per block
  int lt = blockIdx.x, dt = blockIdx.y, n = blockIdx.z;
  const bf16* A = Wt + (size_t)(dt * 64) * C;                 // W rows (M=d, K=c)
  const bf16* B = xT + ((size_t)n * Lsrc + lt * 64) * C;      // Bt rows (N=l, K=c)
  v8f acc[4][4] = {};
  gemm_tile<4, 4>(A, C, B, C, C, lane, acc);
  if (mode == 0) {
#pragma unroll
    for (int i = 0; i < 4; ++i)
#pragma unroll
      for (int j = 0; j < 4; ++j) {
        int l = lt * 64 + j * 16 + (lane & 15);
        int dbase = dt * 64 + i * 16 + ((lane >> 4) << 3);
        v8bf o;
#pragma unroll
        for (int r = 0; r < 8; ++r) o[r] = (bf16)acc[i][j][r];
        *(v8bf*)(out + ((size_t)n * Lsrc + l) * D + dbase) = o;
      }
  } else {
#pragma unroll
    for (int i = 0; i < 4; ++i)
#pragma unroll
      for (int j = 0; j < 4; ++j) {
        int p = lt * 64 + j * 16 + (lane & 15);
        int dbase = dt * 64 + i * 16 + ((lane >> 4) << 3);
#pragma unroll
        for (int r = 0; r < 8; ++r)
          out[((size_t)n * D + dbase + r) * PB + p] = (bf16)acc[i][j][r];
      }
  }
}

// ---------------- fused attention ----------------
// block: 256 threads (8 waves), 32 rows of L; theta staged in LDS;
// logits (32x32 tiles) + softmax + attn@g^T (32x64 tiles)
__global__ void __launch_bounds__(256, 1)
attn_kernel(const bf16* __restrict__ theta_t, const bf16* __restrict__ phi_t,
            const bf16* __restrict__ g_dp, bf16* __restrict__ out_t) {
  extern __shared__ char smem[];
  bf16*  theta_s = (bf16*)smem;                                   // [32][512] bf16  (32 KB)
  float* logits  = (float*)(smem + 32 * D * 2);                   // [32][784] f32   (~98 KB)
  bf16*  attn    = (bf16*)(smem + 32 * D * 2 + 32 * P * 4);       // [32][832] bf16  (52 KB)
  float* inv_s   = (float*)(smem + 32 * D * 2 + 32 * P * 4 + 32 * PB * 2);

  int tid = threadIdx.x;
  int lane = tid & 31, wave = tid >> 5;
  int n = blockIdx.y;
  int l0 = blockIdx.x * 32;

  // Stage this block's 32 theta rows into LDS (vectorized linear copy).
  {
    const bf16* tsrc = theta_t + ((size_t)n * L + l0) * D;
    for (int i = tid; i < 32 * D / 8; i += 256)
      *(v8bf*)(theta_s + (size_t)i * 8) = *(const v8bf*)(tsrc + (size_t)i * 8);
  }
  __syncthreads();

  // Phase 1: logits[l][p] = (theta[l] . phi[p]) * D^-0.5 ; 26 tasks of 32x32
  for (int pt = wave; pt < PB / 32; pt += 8) {
    const bf16* B = phi_t + ((size_t)n * PB + pt * 32) * D;
    v8f acc[2][2] = {};
    gemm_tile<2, 2>(theta_s, D, B, D, D, lane, acc);
#pragma unroll
    for (int i = 0; i < 2; ++i)
#pragma unroll
      for (int j = 0; j < 2; ++j) {
        int p = pt * 32 + j * 16 + (lane & 15);
        if (p < P) {
          int lbase = i * 16 + ((lane >> 4) << 3);
#pragma unroll
          for (int r = 0; r < 8; ++r)
            logits[(lbase + r) * P + p] = acc[i][j][r] * RSQRT_D;
        }
      }
  }
  __syncthreads();

  // Phase 2: softmax over p; write unnormalized bf16 probs, 1/s folded in later
  {
    int row = tid >> 3, sub = tid & 7;  // 8 threads per row, same wave
    float m = -3.0e38f;
    for (int p = sub; p < P; p += 8) m = fmaxf(m, logits[row * P + p]);
    m = fmaxf(m, __shfl_xor(m, 1, 8));
    m = fmaxf(m, __shfl_xor(m, 2, 8));
    m = fmaxf(m, __shfl_xor(m, 4, 8));
    float s = 0.f;
    for (int p = sub; p < P; p += 8) {
      float e = __expf(logits[row * P + p] - m);
      attn[row * PB + p] = (bf16)e;
      s += e;
    }
    s += __shfl_xor(s, 1, 8);
    s += __shfl_xor(s, 2, 8);
    s += __shfl_xor(s, 4, 8);
    for (int p = P + sub; p < PB; p += 8) attn[row * PB + p] = (bf16)0.f;
    if (sub == 0) inv_s[row] = 1.f / s;
  }
  __syncthreads();

  // Phase 3: out[l][d] = (attn[l] . g[d]) * inv_s[l] ; 8 tasks of 32x64
  for (int dt = wave; dt < D / 64; dt += 8) {
    const bf16* B = g_dp + ((size_t)n * D + dt * 64) * PB;
    v8f acc[2][4] = {};
    gemm_tile<2, 4>(attn, PB, B, PB, PB, lane, acc);
#pragma unroll
    for (int i = 0; i < 2; ++i)
#pragma unroll
      for (int j = 0; j < 4; ++j) {
        int d = dt * 64 + j * 16 + (lane & 15);
        int lbase = i * 16 + ((lane >> 4) << 3);
#pragma unroll
        for (int r = 0; r < 8; ++r) {
          int ll = lbase + r;
          out_t[((size_t)n * L + l0 + ll) * D + d] = (bf16)(acc[i][j][r] * inv_s[ll]);
        }
      }
  }
}

// ---------------- output projection + affine + residual (64x64 tiles) ----------------
__global__ void __launch_bounds__(32, 1)
outproj_kernel(const bf16* __restrict__ Wo, const bf16* __restrict__ out_t,
               const float* __restrict__ x, const float* __restrict__ scale,
               const float* __restrict__ bias, float* __restrict__ y) {
  int lane = threadIdx.x;  // one wave32 per block
  int lt = blockIdx.x, ct = blockIdx.y, n = blockIdx.z;
  const bf16* A = Wo + (size_t)(ct * 64) * D;                  // w_out rows (M=c, K=d)
  const bf16* B = out_t + ((size_t)n * L + lt * 64) * D;       // Bt rows (N=l, K=d)
  v8f acc[4][4] = {};
  gemm_tile<4, 4>(A, D, B, D, D, lane, acc);
#pragma unroll
  for (int i = 0; i < 4; ++i)
#pragma unroll
    for (int j = 0; j < 4; ++j) {
      int l = lt * 64 + j * 16 + (lane & 15);
      int cbase = ct * 64 + i * 16 + ((lane >> 4) << 3);
#pragma unroll
      for (int r = 0; r < 8; ++r) {
        int c = cbase + r;
        size_t idx = ((size_t)n * C + c) * L + l;
        y[idx] = x[idx] + acc[i][j][r] * scale[c] + bias[c];
      }
    }
}

// ---------------- launcher ----------------

extern "C" void kernel_launch(void* const* d_in, const int* in_sizes, int n_in,
                              void* d_out, int out_size, void* d_ws, size_t ws_size,
                              hipStream_t stream) {
  const float* x       = (const float*)d_in[0];
  const float* w_theta = (const float*)d_in[1];
  const float* w_phi   = (const float*)d_in[2];
  const float* w_g     = (const float*)d_in[3];
  const float* w_out   = (const float*)d_in[4];
  const float* nscale  = (const float*)d_in[5];
  const float* nbias   = (const float*)d_in[6];
  float* y = (float*)d_out;
  (void)in_sizes; (void)n_in; (void)out_size; (void)ws_size;

  char* ws = (char*)d_ws;
  size_t off = 0;
  auto alloc = [&](size_t bytes) {
    char* p = ws + off;
    off = (off + bytes + 255) & ~(size_t)255;
    return p;
  };
  bf16* xT      = (bf16*)alloc((size_t)NB * L * C * 2);   // 51.4 MB
  bf16* xpT     = (bf16*)alloc((size_t)NB * PB * C * 2);  // 13.6 MB (pad rows zeroed)
  bf16* wtb     = (bf16*)alloc((size_t)D * C * 2);
  bf16* wpb     = (bf16*)alloc((size_t)D * C * 2);
  bf16* wgb     = (bf16*)alloc((size_t)D * C * 2);
  bf16* wob     = (bf16*)alloc((size_t)C * D * 2);
  bf16* theta_t = (bf16*)alloc((size_t)NB * L * D * 2);   // 25.7 MB  [l][d]
  bf16* phi_t   = (bf16*)alloc((size_t)NB * PB * D * 2);  // 6.8 MB   [p][d]
  bf16* g_dp    = (bf16*)alloc((size_t)NB * D * PB * 2);  // 6.8 MB   [d][p]
  bf16* out_t   = (bf16*)alloc((size_t)NB * L * D * 2);   // 25.7 MB  [l][d]

  // pack weights to bf16
  cvt_bf16_kernel<<<dim3((D * C + 255) / 256), 256, 0, stream>>>(w_theta, wtb, D * C);
  cvt_bf16_kernel<<<dim3((D * C + 255) / 256), 256, 0, stream>>>(w_phi, wpb, D * C);
  cvt_bf16_kernel<<<dim3((D * C + 255) / 256), 256, 0, stream>>>(w_g, wgb, D * C);
  cvt_bf16_kernel<<<dim3((C * D + 255) / 256), 256, 0, stream>>>(w_out, wob, C * D);

  // transpose x (and pooled x, zero-padded to PB rows) to [spatial][channel] bf16
  xpose_kernel<<<dim3(L / 32, C / 32, NB), dim3(32, 8), 0, stream>>>(x, xT);
  pool_xpose_kernel<<<dim3(PB / 32, C / 32, NB), dim3(32, 8), 0, stream>>>(x, xpT);

  // projections (64x64 wave tiles)
  proj_kernel<<<dim3(L / 64, D / 64, NB), 32, 0, stream>>>(wtb, xT, theta_t, L, 0);
  proj_kernel<<<dim3(PB / 64, D / 64, NB), 32, 0, stream>>>(wpb, xpT, phi_t, PB, 0);
  proj_kernel<<<dim3(PB / 64, D / 64, NB), 32, 0, stream>>>(wgb, xpT, g_dp, PB, 1);

  // fused attention (theta staged in LDS; logits + softmax + attn@g^T)
  size_t smem = (size_t)32 * D * 2 + (size_t)32 * P * 4 + (size_t)32 * PB * 2 + 32 * 4;
  attn_kernel<<<dim3(L / 32, NB), 256, smem, stream>>>(theta_t, phi_t, g_dp, out_t);

  // output projection + frozen-BN affine + residual (64x64 tiles)
  outproj_kernel<<<dim3(L / 64, C / 64, NB), 32, 0, stream>>>(wob, out_t, x, nscale, nbias, y);
}
